// ChannelAttention_91319594648309
// MI455X (gfx1250) — matrix operand order repaired and checked by reading, not compiled, not run on hardware
//
#include <hip/hip_runtime.h>

typedef __attribute__((ext_vector_type(16))) _Float16     v16h;
typedef __attribute__((ext_vector_type(8)))  _Float16     v8h;
typedef __attribute__((ext_vector_type(2)))  _Float16     v2h;
typedef __attribute__((ext_vector_type(8)))  float        v8f;
typedef __attribute__((ext_vector_type(4)))  unsigned int v4u;
typedef __attribute__((ext_vector_type(8)))  unsigned int v8u;

#define LDS_STRIDE 40  // 32 K-halfs + 8 pad -> 80B row pitch, conflict-free b128 reads

// ---------------------------------------------------------------------------
// LDS -> WMMA fragment loaders (wave32, v_wmma_f32_16x16x32_f16 layouts)
// A (16x32, MxK): lanes 0-15 M=0-15, elems 0-7 = K[kh*8 .. +7], elems 8-15 = K[16+kh*8 .. +7], kh = lane>>4
// B (32x16, KxN): lane holds N = lane&15, elems 0-7 = K[kh*8 .. +7], elems 8-15 = K[16+kh*8 .. +7]; LDS staged as [N][K]
// ---------------------------------------------------------------------------
__device__ __forceinline__ v16h load_frag_a(const _Float16* s, int rowBase, int lane) {
    const _Float16* p = s + (rowBase + (lane & 15)) * LDS_STRIDE + (lane >> 4) * 8;
    v8h lo = *(const v8h*)(p);
    v8h hi = *(const v8h*)(p + 16);
    return __builtin_shufflevector(lo, hi, 0,1,2,3,4,5,6,7,8,9,10,11,12,13,14,15);
}
__device__ __forceinline__ v16h load_frag_b(const _Float16* s, int colBase, int lane) {
    const _Float16* p = s + (colBase + (lane & 15)) * LDS_STRIDE + (lane >> 4) * 8;
    v8h lo = *(const v8h*)(p);
    v8h hi = *(const v8h*)(p + 16);
    return __builtin_shufflevector(lo, hi, 0,1,2,3,4,5,6,7,8,9,10,11,12,13,14,15);
}

// ---------------------------------------------------------------------------
// TDM: DMA a (rows x 32) f16 tile (row stride ldaElems halfs) from global into
// LDS with a 16B pad after every 64B row -> row pitch = LDS_STRIDE halfs.
// 2D descriptor, groups 0-1 only (ISA 08_async_tensor 8.3/8.4). Uniform SGPRs.
// ---------------------------------------------------------------------------
__device__ __forceinline__ void tdm_load_tile_f16(const _Float16* gsrc, unsigned int ldsByteOff,
                                                  int ldaElems, int totalRows, int tileRows)
{
    const unsigned long long ga = (unsigned long long)(uintptr_t)gsrc;
    const v4u g0 = {
        1u,                                               // count=1, is_restore=0, gather=0
        ldsByteOff,                                       // lds_addr (bytes)
        (unsigned int)ga,                                 // global_addr[31:0]
        ((unsigned int)(ga >> 32) & 0x01FFFFFFu) | 0x80000000u  // addr[56:32] | type=2
    };
    const unsigned int dim0 = (unsigned int)ldaElems;     // tensor width (elements)
    const unsigned int dim1 = (unsigned int)totalRows;    // tensor height (rows)
    const v8u g1 = {
        (1u << 16) | (1u << 20) | (3u << 22) | (3u << 25),// data_size=2B, pad_en, intvl=16DW, amt=4DW
        (dim0 & 0xFFFFu) << 16,                           // [47:32]=atomic_addr=0, [63:48]=dim0.lo
        (dim0 >> 16) | ((dim1 & 0xFFFFu) << 16),          // dim0.hi | dim1.lo
        (dim1 >> 16) | (32u << 16),                       // dim1.hi | tile_dim0=32
        (unsigned int)tileRows,                           // tile_dim1=rows, tile_dim2=0
        (unsigned int)ldaElems,                           // tensor_dim0_stride lo (elements)
        0u,                                               // stride hi | dim1_stride lo
        0u
    };
    asm volatile("tensor_load_to_lds %0, %1" :: "s"(g0), "s"(g1) : "memory");
}

// ---------------------------------------------------------------------------
// Generic tiled WMMA GEMM: C[M,N] = A[M,K] @ B[K,N] (+bias)
// 256 threads = 8 waves arranged 4(M) x 2(N); wave tile (BM/4) x (BN/2).
// A,B converted to f16 while staged into LDS; f32 accumulate.
// ATDM: A is f16 and dense -> stage the A tile with the Tensor Data Mover
//       (wave 0 issues, TENSORcnt drained before the barrier) while all
//       threads convert/stage the B tile.
// MODE 0: plain store (TO = float or _Float16)
// MODE 1: batched head-mix GEMM with the reference reshape permutation.
// ---------------------------------------------------------------------------
template<int BM, int BN, typename TA, typename TB, typename TO,
         bool BIAS, bool BTRANS, int MODE, bool ATDM>
__global__ __launch_bounds__(256)
void wmma_gemm(const TA* __restrict__ A, const TB* __restrict__ Bm,
               const float* __restrict__ bias, TO* __restrict__ Cm,
               int M, int N, int K, int lda, int ldb, int ldc)
{
    constexpr int BK = 32;
    constexpr int WM = BM / 4;
    constexpr int WN = BN / 2;
    constexpr int TM = WM / 16;
    constexpr int TN = WN / 16;

    __shared__ _Float16 sA[BM * LDS_STRIDE];
    __shared__ _Float16 sB[BN * LDS_STRIDE];

    const int tid   = threadIdx.x;
    const int lane  = tid & 31;
    const int wave  = tid >> 5;
    const int waveM = wave & 3;
    const int waveN = wave >> 2;

    const int nTile = blockIdx.x * BN;
    const int mTile = blockIdx.y * BM;

    const TA* aPtr = A;
    const TB* bPtr = Bm;
    int zb = 0, zg = 0, zh = 0;
    if constexpr (MODE == 1) {
        const int z = blockIdx.z;                 // z = b*8 + g*4 + h
        zb = z >> 3; zg = (z >> 2) & 1; zh = z & 3;
        aPtr = A  + (size_t)zb * 4096 * 192 + zg * 96;   // q slice (b, :, g*96+e)
        bPtr = Bm + (size_t)z  * 96 * 96;                // attn[b,g,h] (96x96, row=d)
    }

    v8f acc[TM][TN];
    #pragma unroll
    for (int i = 0; i < TM; ++i)
        #pragma unroll
        for (int j = 0; j < TN; ++j)
            #pragma unroll
            for (int r = 0; r < 8; ++r) acc[i][j][r] = 0.0f;

    for (int k0 = 0; k0 < K; k0 += BK) {
        // ---- stage A tile (BM x 32) as f16, row-major [m][k], pitch 40 ----
        if constexpr (ATDM) {
            if (wave == 0)
                tdm_load_tile_f16((const _Float16*)aPtr + (size_t)mTile * lda + k0,
                                  (unsigned int)(uintptr_t)(void*)&sA[0],
                                  lda, M, BM);
        } else {
            #pragma unroll
            for (int i = 0; i < BM / 16; ++i) {     // BM*16 half-pairs / 256 thr
                const int idx = tid + i * 256;
                const int r   = idx >> 4;
                const int kk  = (idx & 15) * 2;
                const TA* ap  = aPtr + (size_t)(mTile + r) * lda + k0 + kk;
                v2h pk; pk[0] = (_Float16)(float)ap[0]; pk[1] = (_Float16)(float)ap[1];
                *(v2h*)&sA[r * LDS_STRIDE + kk] = pk;
            }
        }
        // ---- stage B tile (32 x BN) as f16, transposed [n][k] ----
        #pragma unroll
        for (int i = 0; i < BN / 16; ++i) {
            const int idx = tid + i * 256;
            const int n   = idx >> 4;
            const int kk  = (idx & 15) * 2;
            float f0, f1;
            if constexpr (BTRANS) {                 // B stored [n][k]
                const TB* bp = bPtr + (size_t)(nTile + n) * ldb + k0 + kk;
                f0 = (float)bp[0]; f1 = (float)bp[1];
            } else {                                // B stored [k][n]
                const TB* bp = bPtr + (size_t)(k0 + kk) * ldb + nTile + n;
                f0 = (float)bp[0]; f1 = (float)bp[ldb];
            }
            v2h pk; pk[0] = (_Float16)f0; pk[1] = (_Float16)f1;
            *(v2h*)&sB[n * LDS_STRIDE + kk] = pk;
        }
        if constexpr (ATDM) __builtin_amdgcn_s_wait_tensorcnt(0);  // no-op for waves 1..7
        __syncthreads();

        v16h af[TM], bf[TN];
        #pragma unroll
        for (int i = 0; i < TM; ++i) af[i] = load_frag_a(sA, waveM * WM + i * 16, lane);
        #pragma unroll
        for (int j = 0; j < TN; ++j) bf[j] = load_frag_b(sB, waveN * WN + j * 16, lane);
        #pragma unroll
        for (int i = 0; i < TM; ++i)
            #pragma unroll
            for (int j = 0; j < TN; ++j)
                acc[i][j] = __builtin_amdgcn_wmma_f32_16x16x32_f16(
                    false, af[i], false, bf[j], (short)0, acc[i][j], false, false);
        __syncthreads();
    }

    // ---- store (C layout: lane N = lane&15, M = vgpr + 8*(lane>>4)) ----
    const int rBase = 8 * (lane >> 4);
    const int cLane = lane & 15;
    #pragma unroll
    for (int i = 0; i < TM; ++i)
        #pragma unroll
        for (int j = 0; j < TN; ++j)
            #pragma unroll
            for (int r = 0; r < 8; ++r) {
                const int row = mTile + waveM * WM + i * 16 + rBase + r;
                const int col = nTile + waveN * WN + j * 16 + cLane;
                float v = acc[i][j][r];
                if constexpr (BIAS) v += bias[col];
                if constexpr (MODE == 1) {
                    // out(b,G,HG,n,HD) -> transpose(0,2,1,3,4) -> reshape(b,n,c)
                    const int nn = (zh * 2 + zg) * 512 + (row >> 3);
                    const int cc = (row & 7) * 96 + col;
                    Cm[(size_t)zb * 4096 * 768 + (size_t)nn * 768 + cc] = (TO)v;
                } else {
                    Cm[(size_t)row * ldc + col] = (TO)v;
                }
            }
}

// ---------------------------------------------------------------------------
// Channel attention core: attn[b,g,h] += SCALE * K^T V over a 512-token chunk.
// 128 threads = 4 waves in a 2x2 grid of 48x48 output tiles (96x96 total).
// K/V slabs (32 x 96) are transposed into LDS as [d][n] / [e][n].
// ---------------------------------------------------------------------------
__global__ __launch_bounds__(128)
void attn_kv_kernel(const _Float16* __restrict__ kv, float* __restrict__ attnAcc)
{
    __shared__ _Float16 sK[96 * LDS_STRIDE];
    __shared__ _Float16 sV[96 * LDS_STRIDE];

    const int tid  = threadIdx.x;
    const int lane = tid & 31;
    const int wave = tid >> 5;
    const int waveM = wave >> 1;          // 0..1 (d blocks of 48)
    const int waveN = wave & 1;           // 0..1 (e blocks of 48)

    const int z = blockIdx.z;             // b*8 + g*4 + h
    const int b = z >> 3, g = (z >> 2) & 1, h = z & 3;
    const int koff = (g * 4 + h) * 96;    // k cols; v cols at +768
    const _Float16* base = kv + (size_t)b * 4096 * 1536;
    const int nStart = blockIdx.x * 512;

    v8f acc[3][3];
    #pragma unroll
    for (int i = 0; i < 3; ++i)
        #pragma unroll
        for (int j = 0; j < 3; ++j)
            #pragma unroll
            for (int r = 0; r < 8; ++r) acc[i][j][r] = 0.0f;

    for (int n0 = nStart; n0 < nStart + 512; n0 += 32) {
        // 768 tasks: {K,V} x 12 d-octets x 32 n ; one b128 global read each
        #pragma unroll
        for (int i = 0; i < 6; ++i) {
            const int idx = tid + i * 128;
            const int mat = idx >= 384;         // uniform per iteration
            const int r   = idx - (mat ? 384 : 0);
            const int nn  = r & 31;
            const int d8  = r >> 5;             // 0..11
            const _Float16* p = base + (size_t)(n0 + nn) * 1536
                              + (mat ? 768 + koff : koff) + d8 * 8;
            v8h val = *(const v8h*)p;
            _Float16* s = (mat ? sV : sK) + (d8 * 8) * LDS_STRIDE + nn;
            #pragma unroll
            for (int j = 0; j < 8; ++j) s[j * LDS_STRIDE] = val[j];
        }
        __syncthreads();

        v16h af[3], bf[3];
        #pragma unroll
        for (int i = 0; i < 3; ++i) af[i] = load_frag_a(sK, waveM * 48 + i * 16, lane);
        #pragma unroll
        for (int j = 0; j < 3; ++j) bf[j] = load_frag_b(sV, waveN * 48 + j * 16, lane);
        #pragma unroll
        for (int i = 0; i < 3; ++i)
            #pragma unroll
            for (int j = 0; j < 3; ++j)
                acc[i][j] = __builtin_amdgcn_wmma_f32_16x16x32_f16(
                    false, af[i], false, bf[j], (short)0, acc[i][j], false, false);
        __syncthreads();
    }

    const float SCALE = 0.10206207261596575f;   // 96^-0.5
    float* out = attnAcc + (size_t)z * 96 * 96;
    const int rBase = 8 * (lane >> 4);
    const int cLane = lane & 15;
    #pragma unroll
    for (int i = 0; i < 3; ++i)
        #pragma unroll
        for (int j = 0; j < 3; ++j)
            #pragma unroll
            for (int r = 0; r < 8; ++r) {
                const int d = waveM * 48 + i * 16 + rBase + r;
                const int e = waveN * 48 + j * 16 + cLane;
                atomicAdd(&out[d * 96 + e], acc[i][j][r] * SCALE);
            }
}

// ---------------------------------------------------------------------------
__global__ void zero_kernel(float* __restrict__ p, int n)
{
    const int i = blockIdx.x * 256 + threadIdx.x;
    if (i < n) p[i] = 0.0f;
}

// softmax over e (last axis, length 96) of attn[64][96][96]; output f16
__global__ void softmax_kernel(const float* __restrict__ attnAcc,
                               _Float16* __restrict__ attnH)
{
    const int z = blockIdx.x;           // 64
    const int d = threadIdx.x;          // 96
    const float* row = attnAcc + (size_t)z * 9216 + d * 96;
    _Float16*   orow = attnH   + (size_t)z * 9216 + d * 96;
    float m = -3.402823466e38f;
    for (int e = 0; e < 96; ++e) m = fmaxf(m, row[e]);
    float s = 0.0f;
    for (int e = 0; e < 96; ++e) s += __expf(row[e] - m);
    const float inv = 1.0f / s;
    for (int e = 0; e < 96; ++e) orow[e] = (_Float16)(__expf(row[e] - m) * inv);
}

// ---------------------------------------------------------------------------
extern "C" void kernel_launch(void* const* d_in, const int* in_sizes, int n_in,
                              void* d_out, int out_size, void* d_ws, size_t ws_size,
                              hipStream_t stream)
{
    const float* x     = (const float*)d_in[0];   // (8,4096,768)
    const float* Wq    = (const float*)d_in[1];   // (768,192)
    const float* Wkv   = (const float*)d_in[2];   // (768,1536)
    const float* Wproj = (const float*)d_in[3];   // (768,768)
    const float* bproj = (const float*)d_in[4];   // (768,)
    float* out = (float*)d_out;                   // (8,4096,768) f32

    // workspace carve (256B aligned)
    char* ws = (char*)d_ws;
    size_t off = 0;
    auto carve = [&](size_t bytes) { char* p = ws + off; off = (off + bytes + 255) & ~(size_t)255; return p; };
    _Float16* q_h     = (_Float16*)carve((size_t)8 * 4096 * 192  * 2);  // 12.6 MB
    _Float16* kv_h    = (_Float16*)carve((size_t)8 * 4096 * 1536 * 2);  // 100.7 MB
    float*    attnAcc = (float*)   carve((size_t)64 * 96 * 96   * 4);   // 2.36 MB
    _Float16* attn_h  = (_Float16*)carve((size_t)64 * 96 * 96   * 2);   // 1.18 MB
    _Float16* att_out = (_Float16*)carve((size_t)8 * 4096 * 768 * 2);   // 50.3 MB
    (void)ws_size; (void)in_sizes; (void)n_in; (void)out_size;

    // 1) q = x @ Wq  (32768x768 @ 768x192 -> f16); fp32 A -> VALU conversion staging
    wmma_gemm<128, 64, float, float, _Float16, false, false, 0, false>
        <<<dim3(192 / 64, 32768 / 128), 256, 0, stream>>>(
            x, Wq, nullptr, q_h, 32768, 192, 768, 768, 192, 192);

    // 2) kv = x @ Wkv (32768x768 @ 768x1536 -> f16)
    wmma_gemm<128, 128, float, float, _Float16, false, false, 0, false>
        <<<dim3(1536 / 128, 32768 / 128), 256, 0, stream>>>(
            x, Wkv, nullptr, kv_h, 32768, 1536, 768, 768, 1536, 1536);

    // 3) attn = SCALE * K^T V  (64 heads, split over 8 token chunks + f32 atomics)
    const int attnElems = 64 * 96 * 96;
    zero_kernel<<<(attnElems + 255) / 256, 256, 0, stream>>>(attnAcc, attnElems);
    attn_kv_kernel<<<dim3(8, 1, 64), 128, 0, stream>>>(kv_h, attnAcc);

    // 4) softmax over e -> f16
    softmax_kernel<<<64, 96, 0, stream>>>(attnAcc, attn_h);

    // 5) out = Q @ attn^T per head (A tile via TDM), reference reshape permutation store
    wmma_gemm<128, 96, _Float16, _Float16, _Float16, false, true, 1, true>
        <<<dim3(1, 4096 / 128, 64), 256, 0, stream>>>(
            q_h, attn_h, nullptr, att_out, 4096, 96, 96, 192, 96, 768);

    // 6) final = att_out @ Wproj + bproj (A tile via TDM, f32 out)
    wmma_gemm<128, 128, _Float16, float, float, true, false, 0, true>
        <<<dim3(768 / 128, 32768 / 128), 256, 0, stream>>>(
            att_out, Wproj, bproj, out, 32768, 768, 768, 768, 768, 768);
}
